// Block_25778393710892
// MI455X (gfx1250) — matrix operand order repaired and checked by reading, not compiled, not run on hardware
//
#include <hip/hip_runtime.h>

typedef __attribute__((ext_vector_type(16))) _Float16 v16h;
typedef __attribute__((ext_vector_type(8)))  _Float16 v8h;
typedef __attribute__((ext_vector_type(8)))  float    v8f;

#define CH 128

// ---------------- degree / dinv ----------------
__global__ void init_deg_kernel(float* deg, int n) {
  int i = blockIdx.x * blockDim.x + threadIdx.x;
  if (i < n) deg[i] = 1.0f;  // self-loop
}

__global__ void count_deg_kernel(const int* __restrict__ dst, float* deg, int e) {
  int i = blockIdx.x * blockDim.x + threadIdx.x;
  if (i < e) atomicAdd(&deg[dst[i]], 1.0f);
}

__global__ void calc_dinv_kernel(const float* __restrict__ deg, float* dinv, int n) {
  int i = blockIdx.x * blockDim.x + threadIdx.x;
  if (i < n) dinv[i] = rsqrtf(deg[i]);
}

// ---------------- GEMM: H = X @ W  (WMMA f32 <- f16) ----------------
// Block: 256 threads (8 waves). Each block: 16 rows of X, all 128 cols.
// Wave w computes the 16x16 tile at columns [16w, 16w+16).
// W is staged TRANSPOSED in LDS so each B fragment is 2 contiguous b128 loads.
// LDS row stride padded to 136 halves (272 B) -> lanes land 4 banks apart.
template <typename IT>
__global__ void gcn_gemm_kernel(const IT* __restrict__ X, const float* __restrict__ W,
                                float* __restrict__ H, int n) {
  constexpr int LDW = 136;  // padded stride (halves); 272 B, 16 B aligned
  __shared__ alignas(16) _Float16 WhT[CH * LDW];  // W^T: [col][k], ~34 KB
  __shared__ alignas(16) _Float16 Xh[16 * LDW];   // X tile: [row][k], ~4.3 KB
  const int t  = threadIdx.x;
  const int m0 = blockIdx.x * 16;

  // W^T: WhT[c][r] = W[r][c]. Global reads strided (W is 64 KB, L2-hot);
  // LDS writes contiguous per thread walk -> no write conflicts.
  for (int i = t; i < CH * CH; i += 256) {
    int c = i >> 7, r = i & (CH - 1);
    WhT[c * LDW + r] = (_Float16)W[r * CH + c];
  }
  for (int i = t; i < 16 * CH; i += 256) {
    int r = i >> 7, c = i & (CH - 1);
    int row = m0 + r;
    float v = (row < n) ? (float)X[row * CH + c] : 0.0f;
    Xh[r * LDW + c] = (_Float16)v;
  }
  __syncthreads();

  const int wave = t >> 5;    // n-tile 0..7
  const int lane = t & 31;
  const int m    = lane & 15; // A row / B,D column-within-tile
  const int kh   = lane >> 4; // K-half selector
  const int ncol = wave * 16 + m;

  // A 16x32 f16: element (v,j): k = (v/4)*16 + kh*8 + (v%4)*2 + j
  //   -> halves [kh*8 .. kh*8+7] and [16+kh*8 .. 16+kh*8+7]   (2x b128)
  // B 32x16 f16: same k order as A
  //   -> halves [kh*8 .. kh*8+7] and [16+kh*8 .. 16+kh*8+7]   (2x b128)
  const _Float16* arow = Xh  + m    * LDW + kh * 8;
  const _Float16* brow = WhT + ncol * LDW + kh * 8;

  v8f acc = {};
#pragma unroll
  for (int kt = 0; kt < 4; ++kt) {  // K = 4 * 32
    v8h a0 = *(const v8h*)(arow + kt * 32);
    v8h a1 = *(const v8h*)(arow + kt * 32 + 16);
    v8h b0 = *(const v8h*)(brow + kt * 32);
    v8h b1 = *(const v8h*)(brow + kt * 32 + 16);
    v16h a = __builtin_shufflevector(a0, a1, 0, 1, 2, 3, 4, 5, 6, 7,
                                     8, 9, 10, 11, 12, 13, 14, 15);
    v16h b = __builtin_shufflevector(b0, b1, 0, 1, 2, 3, 4, 5, 6, 7,
                                     8, 9, 10, 11, 12, 13, 14, 15);
    acc = __builtin_amdgcn_wmma_f32_16x16x32_f16(
        /*neg_a=*/false, a, /*neg_b=*/false, b,
        /*c_mod=*/(short)0, acc, /*reuse_a=*/false, /*reuse_b=*/false);
  }

  // D 16x16 f32: lane%16 = N, M = r + 8*(lane/16)
  // Block-uniform bound check: straight-line stores on the fast path
  // (no per-row exec-mask churn; tail block takes the guarded loop).
  const int mbase = m0 + (lane >> 4) * 8;
  if (m0 + 16 <= n) {
#pragma unroll
    for (int r = 0; r < 8; ++r) H[(mbase + r) * CH + ncol] = acc[r];
  } else {
#pragma unroll
    for (int r = 0; r < 8; ++r) {
      int row = mbase + r;
      if (row < n) H[row * CH + ncol] = acc[r];
    }
  }
}

// ---------------- agg init: self-loop term + bias ----------------
__global__ void init_agg_kernel(const float* __restrict__ H, const float* __restrict__ dinv,
                                const float* __restrict__ bias, float* __restrict__ agg, int n) {
  int id = blockIdx.x * blockDim.x + threadIdx.x;
  if (id >= n * CH) return;
  int row = id >> 7, c = id & (CH - 1);
  float di = dinv[row];
  agg[id] = H[id] * di * di + bias[c];
}

// ---------------- edge scatter: agg[dst] += h[src] * dinv[src]*dinv[dst] ----------------
__global__ void edge_scatter_kernel(const float* __restrict__ H, const float* __restrict__ dinv,
                                    const int* __restrict__ src, const int* __restrict__ dst,
                                    float* __restrict__ agg, int e) {
  long long id = (long long)blockIdx.x * blockDim.x + threadIdx.x;
  if (id >= (long long)e * CH) return;
  int ed = (int)(id >> 7);
  int c  = (int)(id & (CH - 1));
  int s = src[ed], d = dst[ed];
  float w = dinv[s] * dinv[d];
  atomicAdd(&agg[d * CH + c], H[s * CH + c] * w);  // L2-resident float atomic
}

// ---------------- BatchNorm ----------------
__global__ void zero_stats_kernel(float* sums) {
  sums[threadIdx.x] = 0.0f;  // <<<1,256>>> : sums[0..127]=sum, [128..255]=sumsq
}

__global__ void bn_stats_kernel(const float* __restrict__ A, float* __restrict__ sums, int n) {
  // blockDim = 128 (thread == channel); each block owns 256 rows
  int c  = threadIdx.x;
  int r0 = blockIdx.x * 256;
  int r1 = min(r0 + 256, n);
  float s = 0.0f, sq = 0.0f;
  for (int r = r0; r < r1; ++r) {
    float v = A[r * CH + c];
    s += v; sq += v * v;
  }
  atomicAdd(&sums[c], s);
  atomicAdd(&sums[CH + c], sq);
}

__global__ void bn_finalize_kernel(const float* __restrict__ sums,
                                   const float* __restrict__ gamma, const float* __restrict__ beta,
                                   float* __restrict__ scale, float* __restrict__ shift, int n) {
  int c = threadIdx.x;  // <<<1,128>>>
  float inv_n = 1.0f / (float)n;
  float mu  = sums[c] * inv_n;
  float var = sums[CH + c] * inv_n - mu * mu;
  float rstd = rsqrtf(var + 1e-5f);
  float sc = gamma[c] * rstd;
  scale[c] = sc;
  shift[c] = beta[c] - mu * sc;
}

__global__ void bn_relu_to_half_kernel(const float* __restrict__ A,
                                       const float* __restrict__ scale, const float* __restrict__ shift,
                                       _Float16* __restrict__ out, int n) {
  int id = blockIdx.x * blockDim.x + threadIdx.x;
  if (id >= n * CH) return;
  int c = id & (CH - 1);
  float v = A[id] * scale[c] + shift[c];
  out[id] = (_Float16)fmaxf(v, 0.0f);
}

__global__ void bn_residual_relu_kernel(const float* __restrict__ A,
                                        const float* __restrict__ scale, const float* __restrict__ shift,
                                        const float* __restrict__ X, float* __restrict__ out, int n) {
  int id = blockIdx.x * blockDim.x + threadIdx.x;
  if (id >= n * CH) return;
  int c = id & (CH - 1);
  float v = A[id] * scale[c] + shift[c] + X[id];
  out[id] = fmaxf(v, 0.0f);  // safe in place: same-index read/write
}

// ---------------- launch ----------------
static inline size_t align256(size_t x) { return (x + 255) & ~(size_t)255; }

extern "C" void kernel_launch(void* const* d_in, const int* in_sizes, int n_in,
                              void* d_out, int out_size, void* d_ws, size_t ws_size,
                              hipStream_t stream) {
  const float* x      = (const float*)d_in[0];
  const float* W1     = (const float*)d_in[1];
  const float* b1     = (const float*)d_in[2];
  const float* W2     = (const float*)d_in[3];
  const float* b2     = (const float*)d_in[4];
  const float* gamma2 = (const float*)d_in[5];
  const float* beta2  = (const float*)d_in[6];
  const int*   eidx   = (const int*)d_in[7];

  const int n = in_sizes[0] / CH;     // 100000
  const int e = in_sizes[7] / 2;      // 1600000
  const int* src = eidx;
  const int* dst = eidx + e;

  // workspace carve-up
  char* ws = (char*)d_ws;
  size_t off = 0;
  float* deg   = (float*)(ws + off); off = align256(off + (size_t)n * 4);
  float* dinv  = (float*)(ws + off); off = align256(off + (size_t)n * 4);
  float* H     = (float*)(ws + off); off = align256(off + (size_t)n * CH * 4);
  float* agg1  = (float*)(ws + off); off = align256(off + (size_t)n * CH * 4);
  _Float16* x1h = (_Float16*)(ws + off); off = align256(off + (size_t)n * CH * 2);
  float* sums  = (float*)(ws + off); off = align256(off + 256 * 4);
  float* scale = (float*)(ws + off); off = align256(off + CH * 4);
  float* shift = (float*)(ws + off); off = align256(off + CH * 4);
  float* agg2  = (float*)d_out;      // reuse output buffer as agg2

  const int nc = n * CH;
  const long long ec = (long long)e * CH;
  dim3 blk(256);
  dim3 gN((n + 255) / 256), gE((e + 255) / 256);
  dim3 gNC((nc + 255) / 256), gEC((unsigned)((ec + 255) / 256));
  dim3 gGemm((n + 15) / 16);
  dim3 gStats((n + 255) / 256);

  // degrees (shared by both convs)
  init_deg_kernel<<<gN, blk, 0, stream>>>(deg, n);
  count_deg_kernel<<<gE, blk, 0, stream>>>(dst, deg, e);
  calc_dinv_kernel<<<gN, blk, 0, stream>>>(deg, dinv, n);

  // ---- conv1 + BN + ReLU (-> f16 activations) ----
  gcn_gemm_kernel<float><<<gGemm, blk, 0, stream>>>(x, W1, H, n);
  init_agg_kernel<<<gNC, blk, 0, stream>>>(H, dinv, b1, agg1, n);
  edge_scatter_kernel<<<gEC, blk, 0, stream>>>(H, dinv, src, dst, agg1, e);
  zero_stats_kernel<<<1, 256, 0, stream>>>(sums);
  bn_stats_kernel<<<gStats, 128, 0, stream>>>(agg1, sums, n);
  bn_finalize_kernel<<<1, 128, 0, stream>>>(sums, gamma2, beta2, scale, shift, n);
  bn_relu_to_half_kernel<<<gNC, blk, 0, stream>>>(agg1, scale, shift, x1h, n);

  // ---- conv2 + BN + residual + ReLU ----
  gcn_gemm_kernel<_Float16><<<gGemm, blk, 0, stream>>>(x1h, W2, H, n);
  init_agg_kernel<<<gNC, blk, 0, stream>>>(H, dinv, b2, agg2, n);
  edge_scatter_kernel<<<gEC, blk, 0, stream>>>(H, dinv, src, dst, agg2, e);
  zero_stats_kernel<<<1, 256, 0, stream>>>(sums);
  bn_stats_kernel<<<gStats, 128, 0, stream>>>(agg2, sums, n);
  bn_finalize_kernel<<<1, 128, 0, stream>>>(sums, gamma2, beta2, scale, shift, n);
  bn_residual_relu_kernel<<<gNC, blk, 0, stream>>>(agg2, scale, shift, x, agg2, n);
}
